// CTX_4303557231316
// MI455X (gfx1250) — matrix operand order repaired and checked by reading, not compiled, not run on hardware
//
#include <hip/hip_runtime.h>
#include <hip/hip_bf16.h>

// MI455X / gfx1250, wave32. All matmuls via v_wmma_f32_16x16x32_bf16.
// Fragment maps (ISA 7.12.2, bf16):
//   A (16x32): lane l -> M = l&15 ; elem e -> K = e + 16*(l>=16)
//              => one contiguous 16-elem run: K in [16*lh, +16) (not the ISA table order)
//   B (32x16): lane l -> N = l&15 ; elem e -> K = e + 16*(l>=16)   (same as A; not the ISA table order)
//   C/D f32  : lane l, vgpr j -> M = j + 8*(l>=16), N = l&15       (8 consecutive M)

typedef __attribute__((ext_vector_type(16))) __bf16 v16bf;
typedef __attribute__((ext_vector_type(8)))  __bf16 v8bf;
typedef __attribute__((ext_vector_type(8)))  float  v8f;
typedef __attribute__((ext_vector_type(4)))  unsigned int u32x4;

#define NB 4
#define NQ 75
#define NS 25
#define NC_ 640
#define NP 100
#define NDK 128
#define NWAY 5
#define NSHOT 5
#define PV 500     // P * NSHOT
#define PVP 512    // padded

__device__ __forceinline__ v8f wmma_bf16(v16bf a, v16bf b, v8f c) {
  return __builtin_amdgcn_wmma_f32_16x16x32_bf16(false, a, false, b, (short)0, c,
                                                 false, false);
}

__device__ __forceinline__ unsigned pk2(float a, float b) {
  unsigned short ua = __builtin_bit_cast(unsigned short, (__bf16)a);
  unsigned short ub = __builtin_bit_cast(unsigned short, (__bf16)b);
  return (unsigned)ua | ((unsigned)ub << 16);
}

__device__ __forceinline__ v16bf zero16() {
  v16bf z;
#pragma unroll
  for (int e = 0; e < 16; ++e) z[e] = (__bf16)0.f;
  return z;
}

// concat two contiguous 16B halves into a 32B A-fragment
__device__ __forceinline__ v16bf frag16(const __bf16* lo, const __bf16* hi) {
  v8bf l = *(const v8bf*)lo;
  v8bf h = *(const v8bf*)hi;
  return __builtin_shufflevector(l, h, 0, 1, 2, 3, 4, 5, 6, 7, 8, 9, 10, 11, 12,
                                 13, 14, 15);
}

// ---------------------------------------------------------------------------
// Projection: out[128 x 100] = W[128 x 640] @ X[640 x 100] per batch (block).
// 8 waves: wave w owns M rows [16w,16w+16), 7 N-tiles (112 padded cols).
// mode 0: q  -> [p*128 + d]               (A-frag friendly: d contiguous)
// mode 1: k  -> [(p*5+sh)*128 + d]        (B-frag friendly: d contiguous)
// mode 2: v  -> [d*512 + p*5+sh] + zero pad cols 500..511
// mode 3: fused qv flat-view reduction (qvSum, bn2)
// ---------------------------------------------------------------------------
__global__ __launch_bounds__(256) void proj_kernel(
    const float* __restrict__ X, const float* __restrict__ W,
    __bf16* __restrict__ outb, float* __restrict__ qvSum,
    float* __restrict__ bn2, int mode) {
  __shared__ v16bf Wf[8][32];     // A fragments, 8 KB
  __shared__ __bf16 Xl[32][112];  // coalesced-stage buffer, 7 KB
  __shared__ v16bf Xf[7][32];     // B fragments, 7 KB
  __shared__ float qvs[128];
  __shared__ float bn2s;

  const int tid = threadIdx.x;
  const int lane = tid & 31;
  const int wave = tid >> 5;
  const int batch = blockIdx.x;
  const float* Xb = X + (size_t)batch * (NC_ * NP);

  if (mode == 3) {
    if (tid < 128) qvs[tid] = 0.f;
    if (tid == 0) bn2s = 0.f;
  }

  v8f zz = {0.f, 0.f, 0.f, 0.f, 0.f, 0.f, 0.f, 0.f};
  v8f acc[7];
  for (int t = 0; t < 7; ++t) acc[t] = zz;

  for (int kc = 0; kc < NC_; kc += 32) {
    __syncthreads();
    // stage W directly in A-fragment layout: unit = (wt, lane, half)
#pragma unroll
    for (int r = 0; r < 2; ++r) {
      int u = tid + r * 256;  // 0..511
      int half = u & 1, l2 = (u >> 1) & 31, wt = u >> 6;
      int row = wt * 16 + (l2 & 15);
      int kb = kc + ((l2 >> 4) & 1) * 16 + half * 8;  // 8 consecutive K
      const float* s = W + row * NC_ + kb;
      u32x4 pk;
      pk.x = pk2(s[0], s[1]);
      pk.y = pk2(s[2], s[3]);
      pk.z = pk2(s[4], s[5]);
      pk.w = pk2(s[6], s[7]);
      ((u32x4*)&Wf[wt][l2])[half] = pk;
    }
    // coalesced global -> bf16 staging of X chunk
    for (int i = tid; i < 32 * 112; i += 256) {
      int c = i / 112, p = i % 112;
      Xl[c][p] = (__bf16)((p < NP) ? Xb[(size_t)(kc + c) * NP + p] : 0.f);
    }
    __syncthreads();
    // repack Xl -> B-fragment layout (K = c fully contiguous per lane)
    for (int u = tid; u < 7 * 32; u += 256) {
      int t = u >> 5, l2 = u & 31;
      int n = t * 16 + (l2 & 15);
      int cb = ((l2 >> 4) & 1) * 16;
      u32x4 lo, hi;
      lo.x = pk2((float)Xl[cb + 0][n], (float)Xl[cb + 1][n]);
      lo.y = pk2((float)Xl[cb + 2][n], (float)Xl[cb + 3][n]);
      lo.z = pk2((float)Xl[cb + 4][n], (float)Xl[cb + 5][n]);
      lo.w = pk2((float)Xl[cb + 6][n], (float)Xl[cb + 7][n]);
      hi.x = pk2((float)Xl[cb + 8][n], (float)Xl[cb + 9][n]);
      hi.y = pk2((float)Xl[cb + 10][n], (float)Xl[cb + 11][n]);
      hi.z = pk2((float)Xl[cb + 12][n], (float)Xl[cb + 13][n]);
      hi.w = pk2((float)Xl[cb + 14][n], (float)Xl[cb + 15][n]);
      ((u32x4*)&Xf[t][l2])[0] = lo;
      ((u32x4*)&Xf[t][l2])[1] = hi;
    }
    __syncthreads();

    v16bf a = Wf[wave][lane];
#pragma unroll
    for (int t = 0; t < 7; ++t) acc[t] = wmma_bf16(a, Xf[t][lane], acc[t]);
  }
  __syncthreads();

  const int dhalf = (lane >> 4) << 3;
  const int d0 = wave * 16 + dhalf;  // 8 consecutive d per lane (C layout)
  if (mode == 0) {
    __bf16* ob = outb + (size_t)batch * (NP * NDK);
    for (int t = 0; t < 7; ++t) {
      int p = t * 16 + (lane & 15);
      if (p < NP) {
        u32x4 pk;
        pk.x = pk2(acc[t][0], acc[t][1]);
        pk.y = pk2(acc[t][2], acc[t][3]);
        pk.z = pk2(acc[t][4], acc[t][5]);
        pk.w = pk2(acc[t][6], acc[t][7]);
        *(u32x4*)(ob + (size_t)p * NDK + d0) = pk;
      }
    }
  } else if (mode == 1) {
    int b = batch / NS, s = batch % NS;
    int n = s / NSHOT, sh = s % NSHOT;
    __bf16* ob = outb + (size_t)(b * NWAY + n) * (PV * NDK);
    for (int t = 0; t < 7; ++t) {
      int p = t * 16 + (lane & 15);
      if (p < NP) {
        u32x4 pk;
        pk.x = pk2(acc[t][0], acc[t][1]);
        pk.y = pk2(acc[t][2], acc[t][3]);
        pk.z = pk2(acc[t][4], acc[t][5]);
        pk.w = pk2(acc[t][6], acc[t][7]);
        *(u32x4*)(ob + (size_t)(p * NSHOT + sh) * NDK + d0) = pk;
      }
    }
  } else if (mode == 2) {
    int b = batch / NS, s = batch % NS;
    int n = s / NSHOT, sh = s % NSHOT;
    __bf16* ob = outb + (size_t)(b * NWAY + n) * (NDK * PVP);
    for (int i = tid; i < 128 * (PVP - PV); i += 256)  // zero pad cols
      ob[(i / (PVP - PV)) * PVP + PV + (i % (PVP - PV))] = (__bf16)0.f;
    for (int t = 0; t < 7; ++t) {
      int p = t * 16 + (lane & 15);
      if (p < NP)
        for (int j = 0; j < 8; ++j)
          ob[(size_t)(d0 + j) * PVP + p * NSHOT + sh] = (__bf16)acc[t][j];
    }
  } else {
    // reference reinterprets [dk,P] as [P,dk] raw memory: flat f = d*100+p,
    // qvSum[f % 128] += val ; bn2 accumulates sum of squares of everything.
    float sq = 0.f;
    for (int t = 0; t < 7; ++t) {
      int p = t * 16 + (lane & 15);
      if (p < NP)
        for (int j = 0; j < 8; ++j) {
          float v = acc[t][j];
          sq += v * v;
          atomicAdd(&qvs[((d0 + j) * NP + p) & 127], v);
        }
    }
    atomicAdd(&bn2s, sq);
    __syncthreads();
    if (tid < 128) qvSum[batch * NDK + tid] = qvs[tid];  // block owns batch
    if (tid == 0) bn2[batch] = bn2s;
  }
}

// ---------------------------------------------------------------------------
// Fused attention + distance per (b,q,n). Loops 7 p-strips of 16 rows:
//   S = q^T K (16x512, contraction d=128) -> softmax -> proto = attn V^T
//   -> an2 / ab partials -> logits.
// q[p][d], k[v][d], v[d][512]: every WMMA fragment is a contiguous 32B load.
// ---------------------------------------------------------------------------
__global__ __launch_bounds__(256) void attn_kernel(
    const __bf16* __restrict__ qb, const __bf16* __restrict__ kb,
    const __bf16* __restrict__ vb, const float* __restrict__ qvSum,
    const float* __restrict__ bn2sq, float* __restrict__ out) {
  __shared__ float Sl[16][512];   // scores / attn f32 (32 KB)
  __shared__ v16bf aBf[16][32];   // attn as A fragments (16 KB)
  __shared__ float red[16][17];
  __shared__ float s_an2, s_ab;

  const int tid = threadIdx.x;
  const int lane = tid & 31;
  const int wave = tid >> 5;
  const int lh8 = ((lane >> 4) & 1) * 8;
  const int lh16 = ((lane >> 4) & 1) * 16;
  const int dhalf = (lane >> 4) << 3;

  const int idx = blockIdx.x;
  const int n = idx % NWAY;
  const int bq = idx / NWAY;
  const int b = bq / NQ;
  const __bf16* qbb = qb + (size_t)bq * (NP * NDK);
  const __bf16* kbb = kb + (size_t)(b * NWAY + n) * (PV * NDK);
  const __bf16* vbb = vb + (size_t)(b * NWAY + n) * (NDK * PVP);

  const float scale = 0.08838834764831845f;  // 1/sqrt(128)
  const float qsl = qvSum[bq * NDK + wave * 16 + (lane & 15)];
  const int dcol = wave * 16 + (lane & 15);  // this wave's proto d-column
  float an2a = 0.f, aba = 0.f;
  if (tid == 0) { s_an2 = 0.f; s_ab = 0.f; }

  v8f zz = {0.f, 0.f, 0.f, 0.f, 0.f, 0.f, 0.f, 0.f};
  const v16bf zb = zero16();

  for (int ps = 0; ps < 7; ++ps) {
    // -- S = q^T K : A from q[p][d] (2x16B), B from k[v][d] (1x32B) --------
    const int p = ps * 16 + (lane & 15);
    v8f sacc[4];
    for (int tt = 0; tt < 4; ++tt) sacc[tt] = zz;
#pragma unroll
    for (int dc = 0; dc < 4; ++dc) {
      v16bf a = zb;
      if (p < NP) {
        const __bf16* qp = qbb + (size_t)p * NDK + dc * 32 + lh16;
        a = frag16(qp, qp + 8);
      }
#pragma unroll
      for (int tt = 0; tt < 4; ++tt) {
        const int v = (wave + tt * 8) * 16 + (lane & 15);
        v16bf bb = zb;
        if (v < PV) bb = *(const v16bf*)(kbb + (size_t)v * NDK + dc * 32 + lh16);
        sacc[tt] = wmma_bf16(a, bb, sacc[tt]);
      }
    }
    __syncthreads();  // previous iter's aBf reads / Sl reads complete
    for (int tt = 0; tt < 4; ++tt) {
      const int col = (wave + tt * 8) * 16 + (lane & 15);
      for (int j = 0; j < 8; ++j) Sl[j + dhalf][col] = sacc[tt][j];
    }
    __syncthreads();

    // -- softmax over 500 valid cols per row ------------------------------
    {
      const int row = tid >> 4, c0 = tid & 15;
      float m = -3.0e38f;
      for (int c = c0; c < PV; c += 16) m = fmaxf(m, Sl[row][c]);
      red[row][c0] = m;
      __syncthreads();
      if (c0 == 0) {
        float mm = red[row][0];
        for (int c = 1; c < 16; ++c) mm = fmaxf(mm, red[row][c]);
        red[row][16] = mm;
      }
      __syncthreads();
      m = red[row][16];
      float ssum = 0.f;
      for (int c = c0; c < PV; c += 16) {
        float e = __expf((Sl[row][c] - m) * scale);
        Sl[row][c] = e;
        ssum += e;
      }
      if (c0 < PVP - PV) Sl[row][PV + c0] = 0.f;  // mask padding
      __syncthreads();
      red[row][c0] = ssum;
      __syncthreads();
      if (c0 == 0) {
        float t = 0.f;
        for (int c = 0; c < 16; ++c) t += red[row][c];
        red[row][16] = t;
      }
      __syncthreads();
      const float inv = 1.f / red[row][16];
      for (int c = c0; c < PV; c += 16) Sl[row][c] *= inv;
      __syncthreads();
    }

    // -- repack attn f32 -> bf16 A-fragment layout (packed 16B stores) ----
    for (int u = tid; u < 16 * 32 * 2; u += 256) {
      int half = u & 1, l2 = (u >> 1) & 31, vc = u >> 6;
      int row = l2 & 15;
      int kbv = vc * 32 + ((l2 >> 4) & 1) * 16 + half * 8;
      const float* s = &Sl[row][kbv];
      u32x4 pk;
      pk.x = pk2(s[0], s[1]);
      pk.y = pk2(s[2], s[3]);
      pk.z = pk2(s[4], s[5]);
      pk.w = pk2(s[6], s[7]);
      ((u32x4*)&aBf[vc][l2])[half] = pk;
    }
    __syncthreads();

    // -- proto = attn @ V^T : A = 1 LDS 32B load, B = 1 global 32B load ---
    v8f pacc = zz;
#pragma unroll
    for (int vc = 0; vc < 16; ++vc) {
      v16bf bb = *(const v16bf*)(vbb + (size_t)dcol * PVP + vc * 32 + lh16);
      pacc = wmma_bf16(aBf[vc][lane], bb, pacc);
    }

    // -- fused distance partials ------------------------------------------
    for (int j = 0; j < 8; ++j) {
      int pr = ps * 16 + j + dhalf;
      if (pr < NP) {
        float v = pacc[j];
        an2a += v * v;
        aba += v * qsl;
      }
    }
  }

  __syncthreads();
  atomicAdd(&s_an2, an2a);
  atomicAdd(&s_ab, aba);
  __syncthreads();
  if (tid == 0) {
    float dist = s_an2 * (1.f / NP) + bn2sq[bq] * (1.f / NP)
               - 2.f * s_ab * (1.f / (NP * NP));
    out[idx] = -dist;
  }
}

// ---------------------------------------------------------------------------
extern "C" void kernel_launch(void* const* d_in, const int* in_sizes, int n_in,
                              void* d_out, int out_size, void* d_ws,
                              size_t ws_size, hipStream_t stream) {
  const float* sx = (const float*)d_in[0];  // support_xf [4,25,640,10,10]
  const float* qx = (const float*)d_in[1];  // query_xf   [4,75,640,10,10]
  const float* Wq = (const float*)d_in[2];  // [128,640]
  const float* Wk = (const float*)d_in[3];
  const float* Wv = (const float*)d_in[4];
  float* out = (float*)d_out;               // [300,5]

  char* ws = (char*)d_ws;
  size_t off = 0;
  auto alloc = [&](size_t bytes) -> void* {
    void* p = ws + off;
    off = (off + bytes + 255) & ~(size_t)255;
    return p;
  };
  __bf16* qbf   = (__bf16*)alloc((size_t)NB * NQ * NP * NDK * 2);       // [p][d]
  __bf16* kbf   = (__bf16*)alloc((size_t)NB * NWAY * PV * NDK * 2);     // [v][d]
  __bf16* vbf   = (__bf16*)alloc((size_t)NB * NWAY * NDK * PVP * 2);    // [d][vp]
  float*  qvSum = (float*)alloc((size_t)NB * NQ * NDK * 4);
  float*  bn2   = (float*)alloc((size_t)NB * NQ * 4);

  proj_kernel<<<NB * NQ, 256, 0, stream>>>(qx, Wq, qbf, nullptr, nullptr, 0);
  proj_kernel<<<NB * NS, 256, 0, stream>>>(sx, Wk, kbf, nullptr, nullptr, 1);
  proj_kernel<<<NB * NS, 256, 0, stream>>>(sx, Wv, vbf, nullptr, nullptr, 2);
  proj_kernel<<<NB * NQ, 256, 0, stream>>>(qx, Wv, nullptr, qvSum, bn2, 3);
  attn_kernel<<<NB * NQ * NWAY, 256, 0, stream>>>(qbf, kbf, vbf, qvSum, bn2, out);
}
